// DeepTraceGNN_27504970564016
// MI455X (gfx1250) — matrix operand order repaired and checked by reading, not compiled, not run on hardware
//
#include <hip/hip_runtime.h>
#include <hip/hip_bf16.h>

#define NNODES 20000
#define DEG 32

typedef _Float16 h16;
typedef __attribute__((ext_vector_type(8)))  _Float16 v8h;
typedef __attribute__((ext_vector_type(16))) _Float16 v16h;
typedef __attribute__((ext_vector_type(8)))  float    v8f;
typedef __attribute__((ext_vector_type(4)))  unsigned int u32x4;
typedef __attribute__((ext_vector_type(4)))  int i32x4;
typedef __attribute__((ext_vector_type(8)))  int i32x8;

#define USE_TDM (__has_builtin(__builtin_amdgcn_tensor_load_to_lds) && \
                 __has_builtin(__builtin_amdgcn_s_wait_tensorcnt))

// ---------------- WMMA helpers ----------------

__device__ __forceinline__ v8f wmma_f16(v16h a, v16h b, v8f c) {
  // D = A(16x32 f16) * B(32x16 f16) + C(16x16 f32)
  return __builtin_amdgcn_wmma_f32_16x16x32_f16(false, a, false, b, (short)0, c,
                                                false, false);
}

__device__ __forceinline__ v16h cat8(v8h a, v8h b) {
  return __builtin_shufflevector(a, b, 0, 1, 2, 3, 4, 5, 6, 7, 8, 9, 10, 11, 12,
                                 13, 14, 15);
}

// B tile stored pre-swizzled as 2 planes of (32 lanes x 8 f16): lane-linear,
// conflict-free ds_load_b128.
__device__ __forceinline__ v16h load_btile(const h16* base, int T, int lane) {
  v8h a = *(const v8h*)(base + (T * 2 + 0) * 256 + lane * 8);
  v8h b = *(const v8h*)(base + (T * 2 + 1) * 256 + lane * 8);
  return cat8(a, b);
}

// A operand (16x32 K-chunk) from a row-major f16 row.
// lanes 0-15 need K[0..8)+K[16..24), lanes 16-31 K[8..16)+K[24..32)  (o = 0 or 8)
__device__ __forceinline__ v16h load_arow(const h16* p, int o) {
  v8h a = *(const v8h*)(p + o);
  v8h b = *(const v8h*)(p + 16 + o);
  return cat8(a, b);
}

// Activations: native v_tanh_f32 TRANS op when the toolchain exposes it.
#if __has_builtin(__builtin_amdgcn_tanhf)
#define TANHF(x) __builtin_amdgcn_tanhf(x)
#else
#define TANHF(x) (1.0f - 2.0f * __builtin_amdgcn_rcpf(__expf(2.0f * (x)) + 1.0f))
#endif
__device__ __forceinline__ float tanhx(float x) { return TANHF(x); }
__device__ __forceinline__ float sigm(float x) {
  return 0.5f + 0.5f * TANHF(0.5f * x);  // == 1/(1+exp(-x))
}

__device__ __forceinline__ v8f splat8(float x) {
  v8f r;
#pragma unroll
  for (int i = 0; i < 8; ++i) r[i] = x;
  return r;
}

// ---------------- TDM: 1-D global -> LDS DMA (weights preload) ----------------
#if USE_TDM
__device__ __forceinline__ void tdm_load_1d(const void* gsrc, void* lds_dst,
                                            int n8 /* 8-byte elements, <=4096 */) {
  unsigned long long ga = (unsigned long long)gsrc;
  unsigned int lds = (unsigned int)(unsigned long long)lds_dst;  // flat[31:0] = LDS offset
  u32x4 g0;
  g0[0] = 1u;                                              // count=1 (valid), user mode
  g0[1] = lds;                                             // lds_addr
  g0[2] = (unsigned int)ga;                                // global_addr[31:0]
  g0[3] = (unsigned int)((ga >> 32) & 0x01FFFFFFu) | (2u << 30);  // ga[56:32], type=2
  i32x8 g1;
  g1[0] = (int)(3u << 16);                                 // data_size = 8B, no mask
  g1[1] = (int)((unsigned)(n8 & 0xFFFF) << 16);            // tensor_dim0[15:0]
  g1[2] = (int)(((unsigned)n8 >> 16) | (1u << 16));        // tensor_dim0[31:16], tensor_dim1=1
  g1[3] = (int)((unsigned)(n8 & 0xFFFF) << 16);            // tile_dim0 = n8
  g1[4] = 0;                                               // tile_dim1/2 unused
  g1[5] = n8;                                              // tensor_dim0_stride[31:0]
  g1[6] = 0;
  g1[7] = 0;
  i32x4 z4 = {0, 0, 0, 0};
#if __clang_major__ >= 23
  i32x8 z8 = {0, 0, 0, 0, 0, 0, 0, 0};
  __builtin_amdgcn_tensor_load_to_lds(g0, g1, z4, z4, z8, 0);
#else
  __builtin_amdgcn_tensor_load_to_lds(g0, g1, z4, z4, 0);
#endif
}
#endif

// ---------------- prep kernels ----------------

__global__ void sort_nbr_kernel(const int* __restrict__ nbr, int* __restrict__ out) {
  int n = blockIdx.x * blockDim.x + threadIdx.x;
  if (n >= NNODES) return;
  int a[DEG];
#pragma unroll
  for (int j = 0; j < DEG; ++j) a[j] = nbr[n * DEG + j];
  for (int i = 1; i < DEG; ++i) {
    int key = a[i];
    int j = i - 1;
    while (j >= 0 && a[j] > key) { a[j + 1] = a[j]; --j; }
    a[j + 1] = key;
  }
#pragma unroll
  for (int j = 0; j < DEG; ++j) out[n * DEG + j] = a[j];
}

__global__ void pad_feat_kernel(const float* __restrict__ f, h16* __restrict__ h0) {
  int idx = blockIdx.x * blockDim.x + threadIdx.x;
  if (idx >= NNODES * 32) return;
  int n = idx >> 5, j = idx & 31;
  h0[idx] = (j < 3) ? (h16)f[n * 3 + j] : (h16)0.f;
}

// Convert f32 weight [rows x srcStride] into f16 WMMA B-tiles (two 16B planes
// per tile). Padded-K mapping: k<S -> source col k if k<realA else zero;
// S<=k<S+tailLen -> source col realA+(k-S).
__global__ void swizzle_w_kernel(const float* __restrict__ W, h16* __restrict__ dst,
                                 int nAh, int srcStride, int S, int realA,
                                 int tailLen, int total) {
  int idx = blockIdx.x * blockDim.x + threadIdx.x;
  if (idx >= total) return;
  int e = idx & 7;
  int L = (idx >> 3) & 31;
  int p = (idx >> 8) & 1;
  int T = idx >> 9;
  int g = T / nAh, ah = T - g * nAh;
  int row = g * 16 + (L & 15);
  int k = 32 * ah + 8 * (L >> 4) + p * 16 + e;
  float v = 0.f;
  int src = -1;
  if (k < S) { if (k < realA) src = k; }
  else if (k < S + tailLen) src = realA + (k - S);
  if (src >= 0) v = W[row * srcStride + src];
  dst[idx] = (h16)v;
}

__global__ void bias_sum_kernel(const float* __restrict__ a,
                                const float* __restrict__ b,
                                float* __restrict__ out) {
  int i = threadIdx.x;
  if (i < 256) out[i] = a[i] + b[i];
}

__global__ void out_kernel(const h16* __restrict__ h, const float* __restrict__ Wout,
                           const float* __restrict__ bout, float* __restrict__ out) {
  int n = blockIdx.x * blockDim.x + threadIdx.x;
  if (n >= NNODES) return;
  float s = bout[0];
#pragma unroll
  for (int j = 0; j < 64; ++j) s += (float)h[(size_t)n * 64 + j] * Wout[j];
  out[n] = s;
}

// ---------------- main layer kernel ----------------
// One wave per 16-node tile; 4 waves / 128 threads per block.
// gates[16n x 256] = Ax*Wih + Ah*Whh + b ; LSTM pointwise ; h -> LDS transpose
// -> A layout for next step. Then comb = [self, agg] @ Wl^T + bl, relu.

template <int IN_PAD>
__global__ __launch_bounds__(128) void layer_kernel(
    const h16* __restrict__ hin, h16* __restrict__ hout,
    const int* __restrict__ nbrs, const h16* __restrict__ gWih,
    const h16* __restrict__ gWhh, const h16* __restrict__ gWl,
    const float* __restrict__ gBsum, const float* __restrict__ gBl) {
  static_assert(IN_PAD == 32 || IN_PAD == 64, "bad IN_PAD");
  constexpr int AHI = IN_PAD / 32;   // K-chunks of input
  constexpr int NT_IH = 16 * AHI;    // B tiles in Wih
  constexpr int NT_HH = 32;          // B tiles in Whh (K=64, 16 gate tiles)
  constexpr int AHC = AHI + 2;       // K-chunks of comb
  constexpr int NT_L = 4 * AHC;      // B tiles in Wl

  extern __shared__ __align__(128) char smem[];
  h16* sWih = (h16*)smem;
  h16* sWhh = sWih + NT_IH * 512;
  h16* sWl = sWhh + NT_HH * 512;
  float* sB = (float*)(sWl + NT_L * 512);
  float* sBl = sB + 256;
  h16* sStage = (h16*)(sBl + 64);  // 4 waves * 16 rows * 72 f16 (padded stride)

  const int tid = threadIdx.x;
  const int wave = tid >> 5;
  const int lane = tid & 31;

#if USE_TDM
  if (wave == 0) {  // one DMA per weight array; EXEC-independent, wave-uniform
    tdm_load_1d(gWih, sWih, NT_IH * 128);
    tdm_load_1d(gWhh, sWhh, NT_HH * 128);
    tdm_load_1d(gWl, sWl, NT_L * 128);
    __builtin_amdgcn_s_wait_tensorcnt(0);
  }
#else
  for (int i = tid; i < NT_IH * 64; i += 128)
    ((uint4*)sWih)[i] = ((const uint4*)gWih)[i];
  for (int i = tid; i < NT_HH * 64; i += 128)
    ((uint4*)sWhh)[i] = ((const uint4*)gWhh)[i];
  for (int i = tid; i < NT_L * 64; i += 128)
    ((uint4*)sWl)[i] = ((const uint4*)gWl)[i];
#endif
  for (int i = tid; i < 256; i += 128) sB[i] = gBsum[i];
  if (tid < 64) sBl[tid] = gBl[tid];
  __syncthreads();

  const int lrow = lane & 15;  // node row (A) / column n (B/D)
  const int lhi = lane >> 4;
  const int o = lhi * 8;
  const int tileBase = (blockIdx.x * 4 + wave) * 16;
  if (tileBase >= NNODES) return;
  const int nodeRow = tileBase + lrow;
  const int* nbrRow = nbrs + nodeRow * DEG;
  h16* stage = sStage + wave * (16 * 72);

  v8f cst[4];
  v16h Ah[2];
#pragma unroll
  for (int ct = 0; ct < 4; ++ct)
#pragma unroll
    for (int e = 0; e < 8; ++e) cst[ct][e] = 0.f;
#pragma unroll
  for (int j = 0; j < 16; ++j) {
    Ah[0][j] = (h16)0;
    Ah[1][j] = (h16)0;
  }

  // software-pipelined neighbor gather: Ax holds step t, Axn prefetches t+1
  v16h Ax[AHI];
  {
    const h16* xp = hin + (size_t)nbrRow[0] * IN_PAD;
#pragma unroll
    for (int ah = 0; ah < AHI; ++ah) Ax[ah] = load_arow(xp + 32 * ah, o);
  }

  for (int t = 0; t < DEG; ++t) {
    v16h Axn[AHI];
    if (t + 1 < DEG) {
      const h16* xpn = hin + (size_t)nbrRow[t + 1] * IN_PAD;
      if (t + 2 < DEG)
        __builtin_prefetch(hin + (size_t)nbrRow[t + 2] * IN_PAD, 0, 3);
#pragma unroll
      for (int ah = 0; ah < AHI; ++ah) Axn[ah] = load_arow(xpn + 32 * ah, o);
    }

#pragma unroll
    for (int ct = 0; ct < 4; ++ct) {  // hidden-column tile (16 cols)
      v8f g4[4];
#pragma unroll
      for (int q = 0; q < 4; ++q) {  // gate i,f,g,o
        const int g = q * 4 + ct;    // gate-column tile in [0,16)
        v8f acc = splat8(sB[g * 16 + lrow]);
#pragma unroll
        for (int ah = 0; ah < AHI; ++ah)
          acc = wmma_f16(Ax[ah], load_btile(sWih, g * AHI + ah, lane), acc);
#pragma unroll
        for (int ah = 0; ah < 2; ++ah)
          acc = wmma_f16(Ah[ah], load_btile(sWhh, g * 2 + ah, lane), acc);
        g4[q] = acc;
      }
#pragma unroll
      for (int e = 0; e < 8; ++e) {
        float iv = sigm(g4[0][e]);
        float fv = sigm(g4[1][e]);
        float gv = tanhx(g4[2][e]);
        float ov = sigm(g4[3][e]);
        float cv = fv * cst[ct][e] + iv * gv;
        cst[ct][e] = cv;
        // D layout: row m = e + 8*lhi, col = ct*16 + lrow
        stage[(e + 8 * lhi) * 72 + ct * 16 + lrow] = (h16)(ov * tanhx(cv));
      }
    }
    // transpose D->A via per-wave LDS staging (in-order per wave)
#pragma unroll
    for (int ah = 0; ah < 2; ++ah)
      Ah[ah] = load_arow(stage + lrow * 72 + 32 * ah, o);

    if (t + 1 < DEG) {
#pragma unroll
      for (int ah = 0; ah < AHI; ++ah) Ax[ah] = Axn[ah];
    }
  }

  // comb = [self, agg] @ Wl^T + bl, relu -> hout (f16, stride 64)
  v16h Ac[AHC];
  const h16* xs = hin + (size_t)nodeRow * IN_PAD;
#pragma unroll
  for (int ah = 0; ah < AHI; ++ah) Ac[ah] = load_arow(xs + 32 * ah, o);
  Ac[AHI] = Ah[0];
  Ac[AHI + 1] = Ah[1];

#pragma unroll
  for (int ct = 0; ct < 4; ++ct) {
    v8f acc = splat8(sBl[ct * 16 + lrow]);
#pragma unroll
    for (int ah = 0; ah < AHC; ++ah)
      acc = wmma_f16(Ac[ah], load_btile(sWl, ct * AHC + ah, lane), acc);
#pragma unroll
    for (int e = 0; e < 8; ++e) {
      float v = acc[e] > 0.f ? acc[e] : 0.f;
      hout[(size_t)(tileBase + e + 8 * lhi) * 64 + ct * 16 + lrow] = (h16)v;
    }
  }
}

// ---------------- launch ----------------

extern "C" void kernel_launch(void* const* d_in, const int* in_sizes, int n_in,
                              void* d_out, int out_size, void* d_ws, size_t ws_size,
                              hipStream_t stream) {
  (void)in_sizes; (void)n_in; (void)out_size; (void)ws_size;
  const float* nf = (const float*)d_in[0];
  const int* nbr = (const int*)d_in[1];
  const float *Wih[3], *Whh[3], *bih[3], *bhh[3], *Wl[3], *bl[3];
  for (int l = 0; l < 3; ++l) {
    int b = 2 + 6 * l;
    Wih[l] = (const float*)d_in[b];
    Whh[l] = (const float*)d_in[b + 1];
    bih[l] = (const float*)d_in[b + 2];
    bhh[l] = (const float*)d_in[b + 3];
    Wl[l] = (const float*)d_in[b + 4];
    bl[l] = (const float*)d_in[b + 5];
  }
  const float* Wout = (const float*)d_in[20];
  const float* bout = (const float*)d_in[21];

  char* ws = (char*)d_ws;
  size_t off = 0;
  auto carve = [&](size_t bytes) -> void* {
    void* p = ws + off;
    off = (off + bytes + 255) & ~(size_t)255;
    return p;
  };
  int* nbr_s = (int*)carve((size_t)NNODES * DEG * 4);
  h16* h0 = (h16*)carve((size_t)NNODES * 32 * 2);
  h16* hA = (h16*)carve((size_t)NNODES * 64 * 2);
  h16* hB = (h16*)carve((size_t)NNODES * 64 * 2);
  h16 *WihS[3], *WhhS[3], *WlS[3];
  float* Bsum[3];
  const int wihTot[3] = {16 * 1 * 512, 16 * 2 * 512, 16 * 2 * 512};
  const int wlTot[3] = {4 * 3 * 512, 4 * 4 * 512, 4 * 4 * 512};
  const int whhTot = 32 * 512;
  for (int l = 0; l < 3; ++l) {
    WihS[l] = (h16*)carve((size_t)wihTot[l] * 2);
    WhhS[l] = (h16*)carve((size_t)whhTot * 2);
    WlS[l] = (h16*)carve((size_t)wlTot[l] * 2);
    Bsum[l] = (float*)carve(256 * 4);
  }

  sort_nbr_kernel<<<(NNODES + 255) / 256, 256, 0, stream>>>(nbr, nbr_s);
  pad_feat_kernel<<<(NNODES * 32 + 255) / 256, 256, 0, stream>>>(nf, h0);

  // layer 0: in_d=3 padded to 32
  swizzle_w_kernel<<<(wihTot[0] + 255) / 256, 256, 0, stream>>>(
      Wih[0], WihS[0], 1, 3, 32, 3, 0, wihTot[0]);
  swizzle_w_kernel<<<(wlTot[0] + 255) / 256, 256, 0, stream>>>(
      Wl[0], WlS[0], 3, 67, 32, 3, 64, wlTot[0]);
  // layers 1,2: in_d=64
  for (int l = 1; l < 3; ++l) {
    swizzle_w_kernel<<<(wihTot[l] + 255) / 256, 256, 0, stream>>>(
        Wih[l], WihS[l], 2, 64, 64, 64, 0, wihTot[l]);
    swizzle_w_kernel<<<(wlTot[l] + 255) / 256, 256, 0, stream>>>(
        Wl[l], WlS[l], 4, 128, 64, 64, 64, wlTot[l]);
  }
  for (int l = 0; l < 3; ++l) {
    swizzle_w_kernel<<<(whhTot + 255) / 256, 256, 0, stream>>>(
        Whh[l], WhhS[l], 2, 64, 64, 64, 0, whhTot);
    bias_sum_kernel<<<1, 256, 0, stream>>>(bih[l], bhh[l], Bsum[l]);
  }

  const int lgrid = ((NNODES / 16) + 3) / 4;  // 313 blocks of 4 waves
  const size_t smem32 =
      (size_t)(16 * 512 + 32 * 512 + 12 * 512) * 2 + (256 + 64) * 4 + 4 * 16 * 72 * 2;
  const size_t smem64 =
      (size_t)(32 * 512 + 32 * 512 + 16 * 512) * 2 + (256 + 64) * 4 + 4 * 16 * 72 * 2;

  layer_kernel<32><<<lgrid, 128, smem32, stream>>>(h0, hA, nbr_s, WihS[0],
                                                   WhhS[0], WlS[0], Bsum[0], bl[0]);
  layer_kernel<64><<<lgrid, 128, smem64, stream>>>(hA, hB, nbr_s, WihS[1],
                                                   WhhS[1], WlS[1], Bsum[1], bl[1]);
  layer_kernel<64><<<lgrid, 128, smem64, stream>>>(hB, hA, nbr_s, WihS[2],
                                                   WhhS[2], WlS[2], Bsum[2], bl[2]);

  out_kernel<<<(NNODES + 255) / 256, 256, 0, stream>>>(hA, Wout, bout,
                                                       (float*)d_out);
}
